// RPN_15607911153701
// MI455X (gfx1250) — matrix operand order repaired and checked by reading, not compiled, not run on hardware
//
#include <hip/hip_runtime.h>
#include <hip/hip_bf16.h>

// ---------- CDNA5 WMMA types ----------
typedef __bf16 bf16x16 __attribute__((ext_vector_type(16)));
typedef float  v8f     __attribute__((ext_vector_type(8)));
typedef unsigned int v4u __attribute__((ext_vector_type(4)));
typedef int v4i __attribute__((ext_vector_type(4)));
typedef int v8i __attribute__((ext_vector_type(8)));

struct U128 { unsigned int w[4]; };
union FragBF { bf16x16 v; U128 q[2]; };

#if defined(__HIP_DEVICE_COMPILE__) && __has_builtin(__builtin_amdgcn_tensor_load_to_lds)
#define USE_TDM 1
#else
#define USE_TDM 0
#endif

__device__ __forceinline__ unsigned short f2bf(float f) {
    unsigned int u = __float_as_uint(f);
    unsigned int r = u + 0x7FFFu + ((u >> 16) & 1u);   // round-to-nearest-even
    return (unsigned short)(r >> 16);
}

__device__ __forceinline__ v8f wmma_bf16(const bf16x16 a, const bf16x16 b, const v8f c) {
    return __builtin_amdgcn_wmma_f32_16x16x32_bf16(false, a, false, b, (short)0, c, false, false);
}

// Issue a 2D TDM load: tile1 x tile0 bf16 tile (row stride stride0 elems)
// from global `src` into LDS at byte offset ldsOff, row-major, contiguous.
__device__ __forceinline__ void tdm_load_2d_bf16(
    const unsigned short* src, unsigned int ldsOff,
    unsigned int dim0, unsigned int dim1,
    unsigned int tile0, unsigned int tile1,
    unsigned long long stride0)
{
#if USE_TDM
    unsigned long long ga = (unsigned long long)(size_t)src;
    v4u g0;
    g0[0] = 1u;                                            // count=1, user mode
    g0[1] = ldsOff;                                        // lds_addr (bytes)
    g0[2] = (unsigned int)(ga & 0xFFFFFFFFu);              // global_addr lo
    g0[3] = (unsigned int)((ga >> 32) & 0x01FFFFFFu) | (2u << 30);  // hi | type=2
    v8i g1;
    g1[0] = 0x00010000;                                    // data_size=1 (2 bytes)
    g1[1] = (int)((dim0 & 0xFFFFu) << 16);                 // tensor_dim0[15:0]
    g1[2] = (int)((dim0 >> 16) | ((dim1 & 0xFFFFu) << 16));// dim0[31:16] | dim1[15:0]
    g1[3] = (int)((dim1 >> 16) | (tile0 << 16));           // dim1[31:16] | tile_dim0
    g1[4] = (int)(tile1);                                  // tile_dim1 | tile_dim2=0
    g1[5] = (int)(unsigned int)(stride0 & 0xFFFFFFFFull);  // stride0[31:0]
    g1[6] = (int)(unsigned int)(stride0 >> 32);            // stride0[47:32]
    g1[7] = 0;
    v4i gz = {0, 0, 0, 0};
#if defined(__clang_major__) && (__clang_major__ >= 23)
    v8i gz8 = {0, 0, 0, 0, 0, 0, 0, 0};
    __builtin_amdgcn_tensor_load_to_lds(g0, g1, gz, gz, gz8, 0);
#else
    __builtin_amdgcn_tensor_load_to_lds(g0, g1, gz, gz, 0);
#endif
#else
    (void)src; (void)ldsOff; (void)dim0; (void)dim1; (void)tile0; (void)tile1; (void)stride0;
#endif
}

// =====================================================================
// Kernel 0: f32 -> bf16 bulk convert (n divisible by 4)
// =====================================================================
__global__ __launch_bounds__(256) void cvt_f32_bf16(
    const float* __restrict__ s, unsigned short* __restrict__ d, int n)
{
    int i = (blockIdx.x * 256 + threadIdx.x) * 4;
    if (i >= n) return;
    float4 v = *(const float4*)(s + i);
    d[i + 0] = f2bf(v.x); d[i + 1] = f2bf(v.y);
    d[i + 2] = f2bf(v.z); d[i + 3] = f2bf(v.w);
}

// =====================================================================
// Kernel 1: pack cls(6x512)+bbox(24x512) into 32x512 bf16 + bias[32]
// =====================================================================
__global__ __launch_bounds__(256) void pack1x1(
    const float* __restrict__ Wcls, const float* __restrict__ bcls,
    const float* __restrict__ Wbbox, const float* __restrict__ bbbox,
    unsigned short* __restrict__ Wc, float* __restrict__ bias)
{
    int i = blockIdx.x * 256 + threadIdx.x;
    if (i < 32) bias[i] = (i < 6) ? bcls[i] : (i < 30 ? bbbox[i - 6] : 0.f);
    if (i >= 32 * 512) return;
    int m = i >> 9, k = i & 511;
    float v = (m < 6) ? Wcls[m * 512 + k] : (m < 30 ? Wbbox[(m - 6) * 512 + k] : 0.f);
    Wc[i] = f2bf(v);
}

// =====================================================================
// Kernel 2: 3x3 conv as implicit GEMM, bf16 WMMA, TDM weight staging,
// double-buffered LDS (overlap TDM + im2col gather with WMMA compute).
//   M=512, N=16384, K=9216; block tile 64x64, BK=32; 8 waves, 2 acc/wave
// =====================================================================
__global__ __launch_bounds__(256) void conv3x3_wmma(
    const unsigned short* __restrict__ Xbf,  // bf16 [1024][128][128]
    const unsigned short* __restrict__ Wbf,  // bf16 [512][9216]
    const float* __restrict__ bias,          // [512]
    unsigned short* __restrict__ Y)          // bf16 [512][16384]
{
    __shared__ unsigned short As[2][64 * 32];   // [m][k]  (TDM destination)
    __shared__ unsigned short Bst[2][64 * 32];  // [n][k]  (transposed)

    const int nBase = blockIdx.x * 64;
    const int mBase = blockIdx.y * 64;
    const int t    = threadIdx.x;
    const int lane = t & 31;
    const int wave = t >> 5;
    const int wm   = wave >> 2;              // 0..1
    const int wn   = wave & 3;               // 0..3

    v8f acc0 = {}; v8f acc1 = {};

    const int brow = t >> 3, bcol = (t & 7) * 8;
    const int hImg = nBase >> 7;
    const int wOff = nBase & 127;

    // incremental (ci, r) for k = kb + brow  (no division in the loop)
    int ci = brow / 9;
    int r  = brow - ci * 9;

    const int lrow  = lane & 15;
    const int lhalf = lane >> 4;
    const unsigned int asOff = (unsigned int)(size_t)(void*)&As[0][0];

    auto issueA = [&](int kb, int buf) {
#if USE_TDM
        if (t == 0)
            tdm_load_2d_bf16(Wbf + (size_t)mBase * 9216 + kb, asOff + buf * 4096,
                             /*dim0=*/9216 - kb, /*dim1=*/64,
                             /*tile0=*/32, /*tile1=*/64, /*stride0=*/9216ull);
#else
        const int arow = t >> 2, acol = (t & 3) * 8;
        *(U128*)(&As[buf][arow * 32 + acol]) =
            *(const U128*)(Wbf + (size_t)(mBase + arow) * 9216 + kb + acol);
#endif
    };
    auto waitA = [&]() {
#if USE_TDM
        if (t == 0) __builtin_amdgcn_s_wait_tensorcnt(0);
#endif
    };
    auto stageB = [&](int buf) {               // branchless im2col gather
        int kh = (r >= 6) ? 2 : ((r >= 3) ? 1 : 0);
        int kw = r - 3 * kh;
        int y  = hImg + kh - 1;
        bool yok = ((unsigned)y) < 128u;
        int yc = yok ? y : 0;
        const unsigned short* xrow = Xbf + (size_t)ci * 16384 + (size_t)yc * 128;
        __builtin_prefetch(xrow + 3 * 16384, 0, 0);       // next k-step's row
        int x0 = wOff + bcol + kw - 1;
        unsigned short* bd = &Bst[buf][bcol * 32 + brow];
        #pragma unroll
        for (int i = 0; i < 8; ++i) {
            int x  = x0 + i;
            int xc = x < 0 ? 0 : (x > 127 ? 127 : x);
            unsigned short v = xrow[xc];                  // unconditional load
            if (!yok || x < 0 || x > 127) v = 0;          // cndmask, no branch
            bd[i * 32] = v;
        }
        ci += 3; r += 5; if (r >= 9) { r -= 9; ++ci; }    // advance by 32 = 3*9+5
    };

    // prologue: fill buffer 0
    issueA(0, 0);
    stageB(0);
    waitA();
    __syncthreads();

    int buf = 0;
    for (int kb = 0; kb < 9216; kb += 32) {
        int nxt = buf ^ 1;
        if (kb + 32 < 9216) {                 // stage next while computing
            issueA(kb + 32, nxt);
            stageB(nxt);
        }
        FragBF a0, a1, b;
        const unsigned short* ar0 = &As[buf][((2 * wm) * 16 + lrow) * 32];
        const unsigned short* ar1 = &As[buf][((2 * wm + 1) * 16 + lrow) * 32];
        const unsigned short* rbp = &Bst[buf][(wn * 16 + lrow) * 32 + lhalf * 8];
        a0.q[0] = *(const U128*)(ar0 + lhalf * 8);
        a0.q[1] = *(const U128*)(ar0 + 16 + lhalf * 8);
        a1.q[0] = *(const U128*)(ar1 + lhalf * 8);
        a1.q[1] = *(const U128*)(ar1 + 16 + lhalf * 8);
        b.q[0]  = *(const U128*)(rbp);
        b.q[1]  = *(const U128*)(rbp + 16);

        acc0 = wmma_bf16(a0.v, b.v, acc0);
        acc1 = wmma_bf16(a1.v, b.v, acc1);

        waitA();                              // TDM for nxt done before barrier
        __syncthreads();
        buf = nxt;
    }

    // ---- epilogue: bias + ReLU, streaming bf16 store ----
    const int nIdx = nBase + wn * 16 + lrow;
    #pragma unroll
    for (int rr = 0; rr < 8; ++rr) {
        int m0 = mBase + (2 * wm) * 16 + lhalf * 8 + rr;
        float v0 = acc0[rr] + bias[m0]; v0 = v0 > 0.f ? v0 : 0.f;
        unsigned short b0 = f2bf(v0);
        __builtin_nontemporal_store(b0, &Y[(size_t)m0 * 16384 + nIdx]);
        int m1 = m0 + 16;
        float v1 = acc1[rr] + bias[m1]; v1 = v1 > 0.f ? v1 : 0.f;
        unsigned short b1 = f2bf(v1);
        __builtin_nontemporal_store(b1, &Y[(size_t)m1 * 16384 + nIdx]);
    }
}

// =====================================================================
// Kernel 3: fused 1x1 heads: raw[32][16384] = Wc[32][512] @ conv1 + bias
// Weight block (32 KB) preloaded with a single TDM transfer.
// =====================================================================
__global__ __launch_bounds__(256) void gemm1x1_wmma(
    const unsigned short* __restrict__ Cb,   // bf16 conv1 [512][16384]
    const unsigned short* __restrict__ Wc,   // bf16 [32][512]
    const float* __restrict__ bias32,
    float* __restrict__ raw)                 // [32][16384]
{
    __shared__ unsigned short As[32 * 512];   // whole weight matrix (32 KB)
    __shared__ unsigned short Bst[128 * 32];  // [n][k] (8 KB)

    const int nBase = blockIdx.x * 128;
    const int t    = threadIdx.x;
    const int lane = t & 31;
    const int wave = t >> 5;
    const int mt   = wave & 1;
    const int nt0  = (wave >> 1) * 2;

#if USE_TDM
    if (t == 0) {
        tdm_load_2d_bf16(Wc, (unsigned int)(size_t)(void*)As,
                         /*dim0=*/16384, /*dim1=*/1,
                         /*tile0=*/16384, /*tile1=*/1, /*stride0=*/16384ull);
        __builtin_amdgcn_s_wait_tensorcnt(0);
    }
#else
    for (int i = t; i < 32 * 512; i += 256) As[i] = Wc[i];
#endif
    __syncthreads();

    v8f acc0 = {}; v8f acc1 = {};
    const int lrow  = lane & 15;
    const int lhalf = lane >> 4;
    const unsigned short* rb0 = Bst + (nt0 * 16 + lrow) * 32 + lhalf * 8;
    const unsigned short* rb1 = Bst + ((nt0 + 1) * 16 + lrow) * 32 + lhalf * 8;

    for (int kb = 0; kb < 512; kb += 32) {
        int k  = t >> 3;
        int n0 = (t & 7) * 16;
        const unsigned short* src = Cb + (size_t)(kb + k) * 16384 + nBase + n0;
        #pragma unroll
        for (int i = 0; i < 16; ++i) Bst[(n0 + i) * 32 + k] = src[i];
        __syncthreads();

        FragBF a, b0, b1;
        const unsigned short* ra = As + (mt * 16 + lrow) * 512 + kb;
        a.q[0]  = *(const U128*)(ra + lhalf * 8);
        a.q[1]  = *(const U128*)(ra + 16 + lhalf * 8);
        b0.q[0] = *(const U128*)(rb0);
        b0.q[1] = *(const U128*)(rb0 + 16);
        b1.q[0] = *(const U128*)(rb1);
        b1.q[1] = *(const U128*)(rb1 + 16);

        acc0 = wmma_bf16(a.v, b0.v, acc0);
        acc1 = wmma_bf16(a.v, b1.v, acc1);
        __syncthreads();
    }

    #pragma unroll
    for (int r = 0; r < 8; ++r) {
        int m = mt * 16 + lhalf * 8 + r;
        float bv = bias32[m];
        int n0 = nBase + nt0 * 16 + lrow;
        float o0 = acc0[r] + bv, o1 = acc1[r] + bv;
        __builtin_nontemporal_store(o0, &raw[(size_t)m * 16384 + n0]);
        __builtin_nontemporal_store(o1, &raw[(size_t)m * 16384 + n0 + 16]);
    }
}

// =====================================================================
// Kernel 4: pairwise softmax (a <-> a+3) -> prob, copy bbox rows.
// =====================================================================
__global__ __launch_bounds__(256) void head_post(
    const float* __restrict__ raw, float* __restrict__ prob, float* __restrict__ bbox)
{
    int p = blockIdx.x * 256 + threadIdx.x;
    if (p >= 16384) return;
    #pragma unroll
    for (int a = 0; a < 3; ++a) {
        float s0 = raw[a * 16384 + p];
        float s1 = raw[(a + 3) * 16384 + p];
        float m  = fmaxf(s0, s1);
        float e0 = __expf(s0 - m), e1 = __expf(s1 - m);
        float inv = 1.f / (e0 + e1);
        prob[a * 16384 + p]       = e0 * inv;
        prob[(a + 3) * 16384 + p] = e1 * inv;
    }
    #pragma unroll
    for (int c = 0; c < 24; ++c)
        bbox[c * 16384 + p] = raw[(6 + c) * 16384 + p];
}

// =====================================================================
// Kernel 5: anchor decode + clip.  i = p*6 + a
// =====================================================================
__constant__ float kAnchorDim[6] = {1.f, 2.f, 4.f, 8.f, 16.f, 32.f};

__global__ __launch_bounds__(256) void decode_boxes(
    const float* __restrict__ prob, const float* __restrict__ bbox,
    const float* __restrict__ meta,
    float* __restrict__ boxes, float* __restrict__ scores)
{
    int i = blockIdx.x * 256 + threadIdx.x;
    if (i >= 98304) return;
    int p = i / 6, a = i - p * 6;
    int h = p >> 7, w = p & 127;

    float sz  = kAnchorDim[a] * 16.f;
    float cxa = (w + 0.5f) * 16.f;
    float cya = (h + 0.5f) * 16.f;

    float dx = bbox[(a * 4 + 0) * 16384 + p];
    float dy = bbox[(a * 4 + 1) * 16384 + p];
    float dw = bbox[(a * 4 + 2) * 16384 + p];
    float dh = bbox[(a * 4 + 3) * 16384 + p];

    float pcx = dx * sz + cxa, pcy = dy * sz + cya;
    float pw  = __expf(dw) * sz, ph = __expf(dh) * sz;

    float imH = meta[0], imW = meta[1];
    float x1 = fminf(fmaxf(pcx - 0.5f * pw, 0.f), imW);
    float y1 = fminf(fmaxf(pcy - 0.5f * ph, 0.f), imH);
    float x2 = fminf(fmaxf(pcx + 0.5f * pw, 0.f), imW);
    float y2 = fminf(fmaxf(pcy + 0.5f * ph, 0.f), imH);

    boxes[i * 4 + 0] = x1; boxes[i * 4 + 1] = y1;
    boxes[i * 4 + 2] = x2; boxes[i * 4 + 3] = y2;
    scores[i] = prob[a * 16384 + p];
}

// ---------- top-k by radix-histogram threshold ----------
__device__ __forceinline__ unsigned int okey(float f) {
    unsigned int u = __float_as_uint(f);
    return u ^ ((u & 0x80000000u) ? 0xFFFFFFFFu : 0x80000000u);
}

__global__ __launch_bounds__(256) void init_sel(
    int* __restrict__ hist, int* __restrict__ ctr,
    float* __restrict__ tS, float* __restrict__ tB, int* __restrict__ alive)
{
    int i = blockIdx.x * 256 + threadIdx.x;
    if (i < 4096) hist[i] = 0;
    if (i < 8)    ctr[i] = 0;
    if (i < 6000) {
        tS[i] = -__builtin_inff();
        alive[i] = 1;
        tB[i * 4 + 0] = 0.f; tB[i * 4 + 1] = 0.f;
        tB[i * 4 + 2] = 0.f; tB[i * 4 + 3] = 0.f;
    }
}

__global__ __launch_bounds__(256) void hist_kernel(
    const float* __restrict__ scores, int* __restrict__ hist)
{
    int i = blockIdx.x * 256 + threadIdx.x;
    if (i >= 98304) return;
    atomicAdd(&hist[okey(scores[i]) >> 20], 1);
}

__global__ void find_thresh(const int* __restrict__ hist, int* __restrict__ ctr)
{
    if (threadIdx.x != 0) return;
    int cum = 0, tbin = 0;
    for (int b = 4095; b >= 0; --b) {
        if (cum + hist[b] >= 6000) { tbin = b; break; }
        cum += hist[b];
    }
    ctr[0] = tbin;
    ctr[1] = cum;
    ctr[2] = 6000 - cum;
}

__global__ __launch_bounds__(256) void compact_topk(
    const float* __restrict__ scores, const float* __restrict__ boxes,
    const int* __restrict__ ctr, int* __restrict__ cnts,
    float* __restrict__ tS, float* __restrict__ tB)
{
    int i = blockIdx.x * 256 + threadIdx.x;
    if (i >= 98304) return;
    int bin = (int)(okey(scores[i]) >> 20);
    int T = ctr[0];
    int slot = -1;
    if (bin > T) {
        slot = atomicAdd(&cnts[4], 1);
    } else if (bin == T) {
        int r = atomicAdd(&cnts[5], 1);
        if (r < ctr[2]) slot = ctr[1] + r;
    }
    if (slot >= 0 && slot < 6000) {
        tS[slot] = scores[i];
        tB[slot * 4 + 0] = boxes[i * 4 + 0];
        tB[slot * 4 + 1] = boxes[i * 4 + 1];
        tB[slot * 4 + 2] = boxes[i * 4 + 2];
        tB[slot * 4 + 3] = boxes[i * 4 + 3];
    }
}

// =====================================================================
// Kernel 9: greedy NMS, 300 sequential rounds, single 1024-thread block.
// =====================================================================
__global__ __launch_bounds__(1024) void nms_kernel(
    const float* __restrict__ tS, const float* __restrict__ tB,
    int* __restrict__ alive, float* __restrict__ rois)
{
    __shared__ float sVal[1024];
    __shared__ int   sIdx[1024];
    __shared__ float bj[4];
    __shared__ int   jSel;
    const int t = threadIdx.x;

    for (int it = 0; it < 300; ++it) {
        float best = -__builtin_inff();
        int   bi   = 0x7fffffff;
        for (int i = t; i < 6000; i += 1024) {
            if (alive[i]) {
                float s = tS[i];
                if (s > best || (s == best && i < bi)) { best = s; bi = i; }
            }
        }
        sVal[t] = best; sIdx[t] = bi;
        __syncthreads();
        for (int off = 512; off > 0; off >>= 1) {
            if (t < off) {
                float o = sVal[t + off]; int oi = sIdx[t + off];
                if (o > sVal[t] || (o == sVal[t] && oi < sIdx[t])) { sVal[t] = o; sIdx[t] = oi; }
            }
            __syncthreads();
        }
        if (t == 0) {
            bool valid = sVal[0] > -__builtin_inff();
            int j = valid ? sIdx[0] : -1;
            jSel = j;
            float x1 = 0.f, y1 = 0.f, x2 = 0.f, y2 = 0.f;
            if (valid) { x1 = tB[j*4]; y1 = tB[j*4+1]; x2 = tB[j*4+2]; y2 = tB[j*4+3]; }
            bj[0] = x1; bj[1] = y1; bj[2] = x2; bj[3] = y2;
            rois[it*5+0] = 0.f; rois[it*5+1] = x1; rois[it*5+2] = y1;
            rois[it*5+3] = x2; rois[it*5+4] = y2;
        }
        __syncthreads();
        int j = jSel;
        if (j >= 0) {
            float jx1 = bj[0], jy1 = bj[1], jx2 = bj[2], jy2 = bj[3];
            float areaJ = fmaxf(jx2 - jx1, 0.f) * fmaxf(jy2 - jy1, 0.f);
            for (int i = t; i < 6000; i += 1024) {
                if (!alive[i]) continue;
                float x1 = tB[i*4], y1 = tB[i*4+1], x2 = tB[i*4+2], y2 = tB[i*4+3];
                float iw = fmaxf(fminf(x2, jx2) - fmaxf(x1, jx1), 0.f);
                float ih = fmaxf(fminf(y2, jy2) - fmaxf(y1, jy1), 0.f);
                float inter = iw * ih;
                float area  = fmaxf(x2 - x1, 0.f) * fmaxf(y2 - y1, 0.f);
                float iou   = inter / (area + areaJ - inter + 1e-9f);
                if (iou > 0.7f) alive[i] = 0;
            }
        }
        __syncthreads();
    }
}

// =====================================================================
extern "C" void kernel_launch(void* const* d_in, const int* in_sizes, int n_in,
                              void* d_out, int out_size, void* d_ws, size_t ws_size,
                              hipStream_t stream)
{
    const float* X     = (const float*)d_in[0];
    const float* meta  = (const float*)d_in[1];
    const float* Wconv = (const float*)d_in[3];
    const float* bconv = (const float*)d_in[4];
    const float* Wcls  = (const float*)d_in[5];
    const float* bcls  = (const float*)d_in[6];
    const float* Wbbx  = (const float*)d_in[7];
    const float* bbbx  = (const float*)d_in[8];

    float* out  = (float*)d_out;
    float* rois = out;
    float* prob = out + 1500;
    float* bbox = out + 1500 + 98304;

    char* w = (char*)d_ws;
    size_t off = 0;
    auto carve = [&](size_t bytes) -> void* {
        off = (off + 255) & ~(size_t)255;
        void* p = w + off;
        off += bytes;
        return p;
    };
    unsigned short* Xbf    = (unsigned short*)carve((size_t)1024 * 16384 * 2);
    unsigned short* Wbf    = (unsigned short*)carve((size_t)512 * 9216 * 2);
    unsigned short* conv1  = (unsigned short*)carve((size_t)512 * 16384 * 2);
    unsigned short* WcBf   = (unsigned short*)carve((size_t)32 * 512 * 2);
    float*          bias32 = (float*)carve(32 * sizeof(float));
    float*          raw    = (float*)carve((size_t)32 * 16384 * sizeof(float));
    float*          boxes  = (float*)carve((size_t)98304 * 4 * sizeof(float));
    float*          scores = (float*)carve((size_t)98304 * sizeof(float));
    float*          tB     = (float*)carve((size_t)6000 * 4 * sizeof(float));
    float*          tS     = (float*)carve((size_t)6000 * sizeof(float));
    int*            alive  = (int*)carve((size_t)6000 * sizeof(int));
    int*            hist   = (int*)carve((size_t)4096 * sizeof(int));
    int*            ctr    = (int*)carve((size_t)16 * sizeof(int));
    (void)ws_size; (void)in_sizes; (void)n_in; (void)out_size;

    cvt_f32_bf16<<<(1024 * 16384) / 1024, 256, 0, stream>>>(X, Xbf, 1024 * 16384);
    cvt_f32_bf16<<<(512 * 9216) / 1024, 256, 0, stream>>>(Wconv, Wbf, 512 * 9216);
    pack1x1<<<64, 256, 0, stream>>>(Wcls, bcls, Wbbx, bbbx, WcBf, bias32);
    conv3x3_wmma<<<dim3(256, 8), 256, 0, stream>>>(Xbf, Wbf, bconv, conv1);
    gemm1x1_wmma<<<128, 256, 0, stream>>>(conv1, WcBf, bias32, raw);
    head_post<<<64, 256, 0, stream>>>(raw, prob, bbox);
    decode_boxes<<<384, 256, 0, stream>>>(prob, bbox, meta, boxes, scores);
    init_sel<<<32, 256, 0, stream>>>(hist, ctr, tS, tB, alive);
    hist_kernel<<<384, 256, 0, stream>>>(scores, hist);
    find_thresh<<<1, 32, 0, stream>>>(hist, ctr);
    compact_topk<<<384, 256, 0, stream>>>(scores, boxes, ctr, ctr, tS, tB);
    nms_kernel<<<1, 1024, 0, stream>>>(tS, tB, alive, rois);
}
